// LSTM_01_25589415149940
// MI455X (gfx1250) — matrix operand order repaired and checked by reading, not compiled, not run on hardware
//
#include <hip/hip_runtime.h>

// ---------------- problem constants ----------------
#define B_TOT 4096
#define T_LEN 256
#define IN_F  3
#define H1    32
#define H2    64
#define OUT_F 2

#define G1 (4 * H1)   // 128 gate columns, layer 1
#define G2 (4 * H2)   // 256 gate columns, layer 2
#define K1 64         // padded concat K for layer1: [x(3) | h1(32) | pad]
#define K2 96         // concat K for layer2:        [h1(32) | h2(64)]

#define NWAVES 4
#define NTHREADS (NWAVES * 32)

typedef _Float16 half8  __attribute__((ext_vector_type(8)));
typedef _Float16 half16 __attribute__((ext_vector_type(16)));
typedef float    float8 __attribute__((ext_vector_type(8)));

// ---------------- fast branch-free activations ----------------
__device__ __forceinline__ float fast_rcp(float x) {
    return __builtin_amdgcn_rcpf(x);          // v_rcp_f32
}

#if __has_builtin(__builtin_amdgcn_tanhf)
__device__ __forceinline__ float tanh_(float x) {
    return __builtin_amdgcn_tanhf(x);         // v_tanh_f32 (gfx1250 trans op)
}
#else
__device__ __forceinline__ float tanh_(float x) {
    // 1 - 2/(e^2x + 1): branch-free, saturates to +/-1 for large |x|
    float t = __expf(2.0f * x);               // v_exp_f32
    return 1.0f - 2.0f * fast_rcp(t + 1.0f);
}
#endif

__device__ __forceinline__ float sigmoidf_(float x) {
    // sigma(x) = 0.5 + 0.5*tanh(x/2): single trans op on gfx1250
    return fmaf(0.5f, tanh_(0.5f * x), 0.5f);
}

// A fragment (16x32 f16, M=row across lanes 0-15): lane group g holds the
// contiguous K run [32*s + 16g, 32*s + 16g + 15] of its row.
__device__ __forceinline__ half16 load_A(const _Float16* rowp, int kOff, int g) {
    half8 lo = *(const half8*)(rowp + kOff + 16 * g);
    half8 hi = *(const half8*)(rowp + kOff + 16 * g + 8);
    return __builtin_shufflevector(lo, hi, 0, 1, 2, 3, 4, 5, 6, 7,
                                   8, 9, 10, 11, 12, 13, 14, 15);
}

// B fragment (32x16 f16, N=col across lanes 0-15): lane group g holds the
// contiguous K run [32*s + 16g, 32*s + 16g + 15] of its column.
__device__ __forceinline__ half16 load_B(const _Float16* colp, int kOff, int g) {
    half8 lo = *(const half8*)(colp + kOff + 16 * g);
    half8 hi = *(const half8*)(colp + kOff + 16 * g + 8);
    return __builtin_shufflevector(lo, hi, 0, 1, 2, 3, 4, 5, 6, 7,
                                   8, 9, 10, 11, 12, 13, 14, 15);
}

__global__ __launch_bounds__(NTHREADS)
void lstm2_fused_wmma(const float* __restrict__ x,
                      const float* __restrict__ w_ih1, const float* __restrict__ w_hh1,
                      const float* __restrict__ b_ih1, const float* __restrict__ b_hh1,
                      const float* __restrict__ w_ih2, const float* __restrict__ w_hh2,
                      const float* __restrict__ b_ih2, const float* __restrict__ b_hh2,
                      const float* __restrict__ fc_w, const float* __restrict__ fc_b,
                      float* __restrict__ out)
{
    // f16 weights, gate-column-major (contiguous K per gate column)
    __shared__ __attribute__((aligned(16))) _Float16 W1[G1][K1];   // 16 KB
    __shared__ __attribute__((aligned(16))) _Float16 W2[G2][K2];   // 48 KB
    __shared__ float bias1[G1];
    __shared__ float bias2[G2];
    __shared__ float gates[16 * G2];                               // 16 KB (reused by both layers)
    __shared__ __attribute__((aligned(16))) _Float16 f1[16][K1];   // A staging layer1
    __shared__ __attribute__((aligned(16))) _Float16 f2[16][K2];   // A staging layer2
    __shared__ float c1s[16][H1];
    __shared__ float c2s[16][H2];
    __shared__ float h2f[16][H2];

    const int tid   = threadIdx.x;       // 0..127
    const int wave  = tid >> 5;          // 0..3: gate-tile slice owner
    const int lane  = tid & 31;          // 0..31 (wave32)
    const int g     = lane >> 4;         // half-wave group
    const int n     = lane & 15;         // row (A) / col-in-tile (B,C)
    const int bBase = blockIdx.x * 16;

    // ---------------- one-time init: pack weights to f16 LDS ----------------
    for (int i = tid; i < G1 * K1; i += NTHREADS) {
        int col = i / K1, k = i % K1;
        float v = 0.0f;
        if (k < IN_F)           v = w_ih1[col * IN_F + k];
        else if (k < IN_F + H1) v = w_hh1[col * H1 + (k - IN_F)];
        W1[col][k] = (_Float16)v;
    }
    for (int i = tid; i < G2 * K2; i += NTHREADS) {
        int col = i / K2, k = i % K2;
        float v = (k < H1) ? w_ih2[col * H1 + k] : w_hh2[col * H2 + (k - H1)];
        W2[col][k] = (_Float16)v;
    }
    for (int i = tid; i < G1; i += NTHREADS) bias1[i] = b_ih1[i] + b_hh1[i];
    for (int i = tid; i < G2; i += NTHREADS) bias2[i] = b_ih2[i] + b_hh2[i];
    for (int i = tid; i < 16 * K1; i += NTHREADS) (&f1[0][0])[i] = (_Float16)0.0f;
    for (int i = tid; i < 16 * K2; i += NTHREADS) (&f2[0][0])[i] = (_Float16)0.0f;
    for (int i = tid; i < 16 * H1; i += NTHREADS) (&c1s[0][0])[i] = 0.0f;
    for (int i = tid; i < 16 * H2; i += NTHREADS) (&c2s[0][0])[i] = 0.0f;
    __syncthreads();

    // ---------------- recurrent scan over T ----------------
    for (int t = 0; t < T_LEN; ++t) {
        // stage x_t (3 floats/row) into f1[:, 0:3]
        if (tid < 16) {
            const float* xp = x + ((size_t)(bBase + tid) * T_LEN + t) * IN_F;
            f1[tid][0] = (_Float16)xp[0];
            f1[tid][1] = (_Float16)xp[1];
            f1[tid][2] = (_Float16)xp[2];
        }
        __syncthreads();

        // ---- layer 1 gates: [16 x 128] = f1[16 x 64] @ W1cat^T ----
        // wave w owns N-tiles {2w, 2w+1}; bias folded into elementwise pass.
        #pragma unroll
        for (int q = 0; q < (G1 / 16) / NWAVES; ++q) {
            const int col = (wave * 2 + q) * 16 + n;
            float8 acc = {0.f, 0.f, 0.f, 0.f, 0.f, 0.f, 0.f, 0.f};
            #pragma unroll
            for (int s = 0; s < K1 / 32; ++s) {
                half16 a = load_A(&f1[n][0], 32 * s, g);
                half16 b = load_B(&W1[col][0], 32 * s, g);
                acc = __builtin_amdgcn_wmma_f32_16x16x32_f16(
                          false, a, false, b, (short)0, acc, false, false);
            }
            #pragma unroll
            for (int i = 0; i < 8; ++i)
                gates[(i + 8 * g) * G1 + col] = acc[i];
        }
        __syncthreads();

        // ---- layer 1 elementwise (i,f,g,o) ----
        #pragma unroll
        for (int idx = tid; idx < 16 * H1; idx += NTHREADS) {
            const int r = idx / H1, j = idx % H1;
            float ig = sigmoidf_(gates[r * G1 + j]          + bias1[j]);
            float fg = sigmoidf_(gates[r * G1 + H1 + j]     + bias1[H1 + j]);
            float gg = tanh_(    gates[r * G1 + 2 * H1 + j] + bias1[2 * H1 + j]);
            float og = sigmoidf_(gates[r * G1 + 3 * H1 + j] + bias1[3 * H1 + j]);
            float c  = fg * c1s[r][j] + ig * gg;
            c1s[r][j] = c;
            _Float16 h = (_Float16)(og * tanh_(c));
            f1[r][IN_F + j] = h;   // recurrent input, layer1
            f2[r][j]        = h;   // feed-forward input, layer2
        }
        __syncthreads();

        // ---- layer 2 gates: [16 x 256] = f2[16 x 96] @ W2cat^T ----
        // wave w owns N-tiles {4w .. 4w+3}.
        #pragma unroll
        for (int q = 0; q < (G2 / 16) / NWAVES; ++q) {
            const int col = (wave * 4 + q) * 16 + n;
            float8 acc = {0.f, 0.f, 0.f, 0.f, 0.f, 0.f, 0.f, 0.f};
            #pragma unroll
            for (int s = 0; s < K2 / 32; ++s) {
                half16 a = load_A(&f2[n][0], 32 * s, g);
                half16 b = load_B(&W2[col][0], 32 * s, g);
                acc = __builtin_amdgcn_wmma_f32_16x16x32_f16(
                          false, a, false, b, (short)0, acc, false, false);
            }
            #pragma unroll
            for (int i = 0; i < 8; ++i)
                gates[(i + 8 * g) * G2 + col] = acc[i];
        }
        __syncthreads();

        // ---- layer 2 elementwise ----
        #pragma unroll
        for (int idx = tid; idx < 16 * H2; idx += NTHREADS) {
            const int r = idx / H2, j = idx % H2;
            float ig = sigmoidf_(gates[r * G2 + j]          + bias2[j]);
            float fg = sigmoidf_(gates[r * G2 + H2 + j]     + bias2[H2 + j]);
            float gg = tanh_(    gates[r * G2 + 2 * H2 + j] + bias2[2 * H2 + j]);
            float og = sigmoidf_(gates[r * G2 + 3 * H2 + j] + bias2[3 * H2 + j]);
            float c  = fg * c2s[r][j] + ig * gg;
            c2s[r][j] = c;
            float h  = og * tanh_(c);
            f2[r][H1 + j] = (_Float16)h;  // recurrent input, layer2
            h2f[r][j]     = h;            // fp32 copy for FC
        }
        __syncthreads();
    }

    // ---------------- FC head: out[b, o] = h2_last . fc_w[o] + fc_b[o] ----------------
    if (tid < 16 * OUT_F) {
        const int r = tid & 15;
        const int o = tid >> 4;           // 0..1 == OUT_F
        float acc = fc_b[o];
        #pragma unroll
        for (int j = 0; j < H2; ++j)
            acc += h2f[r][j] * fc_w[o * H2 + j];
        out[(size_t)(bBase + r) * OUT_F + o] = acc;
    }
}

extern "C" void kernel_launch(void* const* d_in, const int* in_sizes, int n_in,
                              void* d_out, int out_size, void* d_ws, size_t ws_size,
                              hipStream_t stream) {
    (void)in_sizes; (void)n_in; (void)d_ws; (void)ws_size; (void)out_size;
    const float* x     = (const float*)d_in[0];
    const float* w_ih1 = (const float*)d_in[1];
    const float* w_hh1 = (const float*)d_in[2];
    const float* b_ih1 = (const float*)d_in[3];
    const float* b_hh1 = (const float*)d_in[4];
    const float* w_ih2 = (const float*)d_in[5];
    const float* w_hh2 = (const float*)d_in[6];
    const float* b_ih2 = (const float*)d_in[7];
    const float* b_hh2 = (const float*)d_in[8];
    const float* fc_w  = (const float*)d_in[9];
    const float* fc_b  = (const float*)d_in[10];
    float* out = (float*)d_out;

    dim3 grid(B_TOT / 16);
    dim3 block(NTHREADS);
    lstm2_fused_wmma<<<grid, block, 0, stream>>>(
        x, w_ih1, w_hh1, b_ih1, b_hh1,
        w_ih2, w_hh2, b_ih2, b_hh2, fc_w, fc_b, out);
}
